// AttentionLayer_25881472926317
// MI455X (gfx1250) — matrix operand order repaired and checked by reading, not compiled, not run on hardware
//
#include <hip/hip_runtime.h>

// ---------------------------------------------------------------------------
// Types for CDNA5 WMMA (wave32): D(16x16 f32) = A(16x32 bf16) x B(32x16 bf16) + C
// ---------------------------------------------------------------------------
typedef __bf16 bf16t;
typedef __attribute__((ext_vector_type(16))) __bf16 v16bf;
typedef __attribute__((ext_vector_type(8)))  __bf16 v8bf;
typedef __attribute__((ext_vector_type(8)))  float   v8f;

#define WMMA_BF16(a, b, c) \
  __builtin_amdgcn_wmma_f32_16x16x32_bf16(false, (a), false, (b), (short)0, (c), false, false)

#define KD    512     // head dim (= reference K)
#define HH    8
#define BB    4
#define TT    2048
#define MROWS (BB * TT)   // 8192
#define NPROJ (KD * HH)   // 4096
#define PSTR  2056        // LDS P row stride in bf16 elems (2048 + 8 pad, 16B aligned)
#define QSCALE 0.2102241038f  // 512^(-0.25)

// A-fragment (16x32 bf16, row-major source): lane holds row = lane&15,
// K = hi*16 + [0..15] contiguous (caller passes base + hi*8).
static __device__ __forceinline__ v16bf ldA(const bf16t* p) {
  union { v16bf v; v8bf h[2]; } u;
  const int ahi = (threadIdx.x & 31) >> 4;
  u.h[0] = *(const v8bf*)(p + ahi * 8);
  u.h[1] = *(const v8bf*)(p + ahi * 8 + 8);
  return u.v;
}
// B-fragment (32x16 bf16): lane holds column = lane&15, K = hi*16 + [0..15]
// contiguous (source must be column-major / pre-transposed).
static __device__ __forceinline__ v16bf ldB(const bf16t* p) {
  return *(const v16bf*)p;
}

// ---------------------------------------------------------------------------
// Elementwise converters
// ---------------------------------------------------------------------------
__global__ void cvt_f32_bf16_kernel(const float* __restrict__ src,
                                    bf16t* __restrict__ dst, int n) {
  int i = blockIdx.x * blockDim.x + threadIdx.x;
  int stride = gridDim.x * blockDim.x;
  for (; i < n; i += stride) dst[i] = (bf16t)src[i];
}

// dst[c*R + r] = src[r*C + c]  (store W transposed so B-fragments are contiguous)
__global__ void cvt_transpose_kernel(const float* __restrict__ src,
                                     bf16t* __restrict__ dst, int R, int C) {
  int i = blockIdx.x * blockDim.x + threadIdx.x;
  int stride = gridDim.x * blockDim.x;
  int n = R * C;
  for (; i < n; i += stride) {
    int r = i / C, c = i - r * C;
    dst[(size_t)c * R + r] = (bf16t)src[i];
  }
}

// ---------------------------------------------------------------------------
// QKV projection: [8192x512]bf16 @ Wt'[4096x512] -> q/k (B,H,T,KD), v^T (B,H,KD,T)
// 4 waves/block, each wave computes a 16x64 strip (4 accumulator tiles).
// gridDim.y selects which of q/k/v.
// ---------------------------------------------------------------------------
__global__ __launch_bounds__(128) void qkv_proj_kernel(
    const bf16t* __restrict__ xb,
    const bf16t* __restrict__ WqT, const bf16t* __restrict__ WkT,
    const bf16t* __restrict__ WvT,
    bf16t* __restrict__ qm, bf16t* __restrict__ km, bf16t* __restrict__ vTm) {
  const int mat  = blockIdx.y;                 // 0=q 1=k 2=v
  const int wave = threadIdx.x >> 5;
  const int lane = threadIdx.x & 31;
  const int col  = lane & 15;
  const int hi   = lane >> 4;
  const int job  = blockIdx.x * 4 + wave;      // 0..32767
  const int m0   = (job >> 6) * 16;            // 512 M-tiles
  const int n0   = (job & 63) * 64;            // 64 N-quads

  const bf16t* WT = (mat == 0) ? WqT : (mat == 1) ? WkT : WvT;
  const bf16t* arow = xb + (size_t)(m0 + col) * KD;
  const bf16t* b0r = WT + (size_t)(n0 +  0 + col) * KD;
  const bf16t* b1r = WT + (size_t)(n0 + 16 + col) * KD;
  const bf16t* b2r = WT + (size_t)(n0 + 32 + col) * KD;
  const bf16t* b3r = WT + (size_t)(n0 + 48 + col) * KD;

  v8f acc[4] = {v8f{}, v8f{}, v8f{}, v8f{}};
#pragma unroll
  for (int kt = 0; kt < KD / 32; ++kt) {
    const int ka = kt * 32 + hi * 8;
    const int kb = kt * 32 + hi * 16;
    v16bf a = ldA(arow + ka);
    acc[0] = WMMA_BF16(a, ldB(b0r + kb), acc[0]);
    acc[1] = WMMA_BF16(a, ldB(b1r + kb), acc[1]);
    acc[2] = WMMA_BF16(a, ldB(b2r + kb), acc[2]);
    acc[3] = WMMA_BF16(a, ldB(b3r + kb), acc[3]);
  }

  const float scale = (mat == 2) ? 1.0f : QSCALE;  // fold K^-1/4 into q and k
#pragma unroll
  for (int j = 0; j < 4; ++j) {
    const int n = n0 + j * 16 + col;
    const int h = n >> 9, kd = n & (KD - 1);
#pragma unroll
    for (int r = 0; r < 8; ++r) {
      const int m = m0 + hi * 8 + r;
      const int b = m >> 11, t = m & (TT - 1);
      const int bh = b * HH + h;
      bf16t val = (bf16t)(acc[j][r] * scale);
      if (mat == 0)      qm[((size_t)bh * TT + t) * KD + kd] = val;
      else if (mat == 1) km[((size_t)bh * TT + t) * KD + kd] = val;
      else               vTm[((size_t)bh * KD + kd) * TT + t] = val;  // transposed
    }
  }
}

// ---------------------------------------------------------------------------
// Attention: one workgroup (4 waves) per (bh, 16-query-row block).
// Two-pass softmax, P (16x2048 bf16) staged in LDS, out = P @ V via WMMA.
// ---------------------------------------------------------------------------
__global__ __launch_bounds__(128) void attn_kernel(
    const bf16t* __restrict__ qm, const bf16t* __restrict__ km,
    const bf16t* __restrict__ vTm, bf16t* __restrict__ attnO) {
  extern __shared__ char smem[];
  bf16t* Pls    = (bf16t*)smem;                       // [16][PSTR]
  float* partial = (float*)(smem + 16 * PSTR * 2);    // [4][16]
  float* mrow    = partial + 64;                      // [16] row max
  float* lrow    = mrow + 16;                         // [16] 1/rowsum

  const int qb = blockIdx.x;          // query block 0..127
  const int bh = blockIdx.y;          // 0..31
  const int b = bh >> 3, h = bh & 7;
  const int wave = threadIdx.x >> 5;
  const int lane = threadIdx.x & 31;
  const int col  = lane & 15;
  const int hi   = lane >> 4;

  const bf16t* qbase = qm  + ((size_t)bh * TT + qb * 16) * KD;
  const bf16t* kbase = km  + (size_t)bh * TT * KD;
  const bf16t* vbase = vTm + (size_t)bh * KD * TT;
  const bf16t* arow  = qbase + (size_t)col * KD;      // A row for this lane

  // ---- pass 1: row maxima of scores ----
  float vmax[8];
#pragma unroll
  for (int r = 0; r < 8; ++r) vmax[r] = -3.0e38f;
  for (int st = wave; st < TT / 16; st += 4) {
    v8f acc = {};
    const bf16t* brow = kbase + (size_t)(st * 16 + col) * KD;  // K^T column = k row
#pragma unroll
    for (int kt = 0; kt < KD / 32; ++kt) {
      acc = WMMA_BF16(ldA(arow + kt * 32 + hi * 8),
                      ldB(brow + kt * 32 + hi * 16), acc);
    }
#pragma unroll
    for (int r = 0; r < 8; ++r) vmax[r] = fmaxf(vmax[r], acc[r]);
  }
#pragma unroll
  for (int r = 0; r < 8; ++r)
    for (int mk = 1; mk <= 8; mk <<= 1)
      vmax[r] = fmaxf(vmax[r], __shfl_xor(vmax[r], mk, 32));
  if (col == 0) {
#pragma unroll
    for (int r = 0; r < 8; ++r) partial[wave * 16 + hi * 8 + r] = vmax[r];
  }
  __syncthreads();
  if (threadIdx.x < 16) {
    float m = partial[threadIdx.x];
    for (int w = 1; w < 4; ++w) m = fmaxf(m, partial[w * 16 + threadIdx.x]);
    mrow[threadIdx.x] = m;
  }
  __syncthreads();

  float rm[8];
#pragma unroll
  for (int r = 0; r < 8; ++r) rm[r] = mrow[hi * 8 + r];

  // ---- pass 2: P = exp(s - m) into LDS, row sums ----
  float vsum[8];
#pragma unroll
  for (int r = 0; r < 8; ++r) vsum[r] = 0.0f;
  for (int st = wave; st < TT / 16; st += 4) {
    v8f acc = {};
    const bf16t* brow = kbase + (size_t)(st * 16 + col) * KD;
#pragma unroll
    for (int kt = 0; kt < KD / 32; ++kt) {
      acc = WMMA_BF16(ldA(arow + kt * 32 + hi * 8),
                      ldB(brow + kt * 32 + hi * 16), acc);
    }
#pragma unroll
    for (int r = 0; r < 8; ++r) {
      float p = __expf(acc[r] - rm[r]);
      vsum[r] += p;
      Pls[(hi * 8 + r) * PSTR + st * 16 + col] = (bf16t)p;
    }
  }
#pragma unroll
  for (int r = 0; r < 8; ++r)
    for (int mk = 1; mk <= 8; mk <<= 1)
      vsum[r] += __shfl_xor(vsum[r], mk, 32);
  if (col == 0) {
#pragma unroll
    for (int r = 0; r < 8; ++r) partial[wave * 16 + hi * 8 + r] = vsum[r];
  }
  __syncthreads();
  if (threadIdx.x < 16) {
    float s = partial[threadIdx.x];
    for (int w = 1; w < 4; ++w) s += partial[w * 16 + threadIdx.x];
    lrow[threadIdx.x] = 1.0f / s;
  }
  __syncthreads();

  float rl[8];
#pragma unroll
  for (int r = 0; r < 8; ++r) rl[r] = lrow[hi * 8 + r];

  // ---- pass 3: out = P @ V ; each wave owns strided 16-col n-tiles ----
  for (int nt = wave; nt < KD / 16; nt += 4) {
    const int n = nt * 16 + col;
    const bf16t* vrow = vbase + (size_t)n * TT;       // V column = v^T row
    const bf16t* prow = Pls + (size_t)col * PSTR;     // P row for this lane
    v8f acc = {};
#pragma unroll 4
    for (int sp = 0; sp < TT / 32; ++sp) {
      acc = WMMA_BF16(ldA(prow + sp * 32 + hi * 8),
                      ldB(vrow + sp * 32 + hi * 16), acc);
    }
#pragma unroll
    for (int r = 0; r < 8; ++r) {
      const int t = qb * 16 + hi * 8 + r;
      attnO[((size_t)(b * TT + t)) * NPROJ + h * KD + n] = (bf16t)(acc[r] * rl[r]);
    }
  }
}

// ---------------------------------------------------------------------------
// Output projection: attnO[8192x4096]bf16 @ Wu + bu -> out[8192x512]f32
// ---------------------------------------------------------------------------
__global__ __launch_bounds__(128) void out_proj_kernel(
    const bf16t* __restrict__ attnO, const bf16t* __restrict__ WuT,
    const float* __restrict__ bu, float* __restrict__ out) {
  const int wave = threadIdx.x >> 5;
  const int lane = threadIdx.x & 31;
  const int col  = lane & 15;
  const int hi   = lane >> 4;
  const int job  = blockIdx.x * 4 + wave;   // 0..4095
  const int m0   = (job >> 3) * 16;         // 512 M-tiles
  const int n0   = (job & 7) * 64;          // 8 N-quads

  const bf16t* arow = attnO + (size_t)(m0 + col) * NPROJ;
  const bf16t* b0r = WuT + (size_t)(n0 +  0 + col) * NPROJ;
  const bf16t* b1r = WuT + (size_t)(n0 + 16 + col) * NPROJ;
  const bf16t* b2r = WuT + (size_t)(n0 + 32 + col) * NPROJ;
  const bf16t* b3r = WuT + (size_t)(n0 + 48 + col) * NPROJ;

  v8f acc[4] = {v8f{}, v8f{}, v8f{}, v8f{}};
#pragma unroll 4
  for (int kt = 0; kt < NPROJ / 32; ++kt) {
    const int ka = kt * 32 + hi * 8;
    const int kb = kt * 32 + hi * 16;
    v16bf a = ldA(arow + ka);
    acc[0] = WMMA_BF16(a, ldB(b0r + kb), acc[0]);
    acc[1] = WMMA_BF16(a, ldB(b1r + kb), acc[1]);
    acc[2] = WMMA_BF16(a, ldB(b2r + kb), acc[2]);
    acc[3] = WMMA_BF16(a, ldB(b3r + kb), acc[3]);
  }
#pragma unroll
  for (int j = 0; j < 4; ++j) {
    const int n = n0 + j * 16 + col;
    const float bias = bu[n];
#pragma unroll
    for (int r = 0; r < 8; ++r) {
      const int m = m0 + hi * 8 + r;
      out[(size_t)m * KD + n] = acc[j][r] + bias;
    }
  }
}

// ---------------------------------------------------------------------------
// Host-side launch
// ---------------------------------------------------------------------------
extern "C" void kernel_launch(void* const* d_in, const int* in_sizes, int n_in,
                              void* d_out, int out_size, void* d_ws, size_t ws_size,
                              hipStream_t stream) {
  const float* x  = (const float*)d_in[0];
  const float* Wq = (const float*)d_in[1];
  const float* Wk = (const float*)d_in[2];
  const float* Wv = (const float*)d_in[3];
  const float* Wu = (const float*)d_in[4];
  const float* bu = (const float*)d_in[5];
  float* out = (float*)d_out;

  char* ws = (char*)d_ws;
  size_t off = 0;
  auto carve = [&](size_t bytes) -> char* {
    char* p = ws + off;
    off += (bytes + 255) & ~(size_t)255;
    return p;
  };
  bf16t* xb   = (bf16t*)carve((size_t)MROWS * KD * 2);      //  8.4 MB
  bf16t* WqT  = (bf16t*)carve((size_t)NPROJ * KD * 2);      //  4.2 MB
  bf16t* WkT  = (bf16t*)carve((size_t)NPROJ * KD * 2);
  bf16t* WvT  = (bf16t*)carve((size_t)NPROJ * KD * 2);
  bf16t* WuT  = (bf16t*)carve((size_t)KD * NPROJ * 2);
  bf16t* qm   = (bf16t*)carve((size_t)BB * HH * TT * KD * 2);  // 67 MB
  bf16t* km   = (bf16t*)carve((size_t)BB * HH * TT * KD * 2);
  bf16t* vTm  = (bf16t*)carve((size_t)BB * HH * KD * TT * 2);
  bf16t* attnO = (bf16t*)carve((size_t)MROWS * NPROJ * 2);     // 67 MB

  // 1) precision conversion + weight transposes
  cvt_f32_bf16_kernel<<<4096, 256, 0, stream>>>(x, xb, MROWS * KD);
  cvt_transpose_kernel<<<2048, 256, 0, stream>>>(Wq, WqT, KD, NPROJ);
  cvt_transpose_kernel<<<2048, 256, 0, stream>>>(Wk, WkT, KD, NPROJ);
  cvt_transpose_kernel<<<2048, 256, 0, stream>>>(Wv, WvT, KD, NPROJ);
  cvt_transpose_kernel<<<2048, 256, 0, stream>>>(Wu, WuT, NPROJ, KD);

  // 2) QKV projection (grid.y selects q/k/v)
  qkv_proj_kernel<<<dim3(8192, 3), 128, 0, stream>>>(xb, WqT, WkT, WvT, qm, km, vTm);

  // 3) attention per (16 query rows, b*H+h); dynamic LDS: P + reductions
  const size_t attn_lds = (size_t)16 * PSTR * 2 + 64 * 4 + 16 * 4 + 16 * 4;
  attn_kernel<<<dim3(TT / 16, BB * HH), 128, attn_lds, stream>>>(qm, km, vTm, attnO);

  // 4) output projection + bias
  out_proj_kernel<<<1024, 128, 0, stream>>>(attnO, WuT, bu, out);
}
